// GVPGNNModel_56822417326607
// MI455X (gfx1250) — matrix operand order repaired and checked by reading, not compiled, not run on hardware
//
#include <hip/hip_runtime.h>

typedef _Float16 half_t;
typedef __attribute__((ext_vector_type(16))) _Float16 v16h;
typedef __attribute__((ext_vector_type(8)))  float    v8f;

#define WEPS 1e-8f

__device__ __forceinline__ float sigm(float x) { return 1.0f / (1.0f + __expf(-x)); }

// ---------------------------------------------------------------------------
// WMMA GEMM: C[M,N] = A[M,K] (f16 row-major) * Bsw[K,N] (f16, fragment-swizzled)
// + bias.  One wave computes a 16x32 C tile (2 accumulators share one A frag).
// K % 32 == 0, N % 32 == 0.
//
// Bsw layout: tiles of 512 halves, tile id = (kk/32)*(N/16) + nt; within a
// tile element (lane*16 + e) is exactly lane `lane`'s e-th v16h element for
// v_wmma_f32_16x16x32_f16 (K = kk + (lane>>4)*8 + (e&7) + (e>>3)*16, col = nt*16 + (lane&15)).
// => B fragment = two contiguous uint4 loads per lane, fully coalesced.
// ---------------------------------------------------------------------------
__global__ void k_gemm(const half_t* __restrict__ A, const half_t* __restrict__ Bsw,
                       const float* __restrict__ bias, float* __restrict__ C,
                       int M, int N, int K) {
  int lane = threadIdx.x & 31;
  int wave = threadIdx.x >> 5;
  int mt = blockIdx.y * 8 + wave;
  int nt2 = blockIdx.x;                       // 32-column tile index
  if (mt * 16 >= M) return;                   // uniform per wave: EXEC stays all-ones
  int g = lane >> 4;                          // lane group (0/1)
  int r = mt * 16 + (lane & 15);              // A row for this lane
  int ntiles = N >> 4;
  const half_t* arow = A + (size_t)r * K + g * 8;
  const half_t* btile = Bsw + ((size_t)nt2 * 2) * 512 + lane * 16;
  size_t bstep = (size_t)ntiles * 512;        // advance one ktile
  v8f acc0 = {};
  v8f acc1 = {};
  for (int kk = 0; kk < K; kk += 32) {
    // A fragment: lanes 0-15 hold K = {0..7,16..23}, lanes 16-31 hold {8..15,24..31}
    union { v16h v; uint4 q[2]; } a, b0, b1;
    const half_t* ap = arow + kk;
    a.q[0] = *(const uint4*)(ap);
    a.q[1] = *(const uint4*)(ap + 16);
    __builtin_prefetch(ap + 32, 0, 1);        // global_prefetch next A k-slice
    const half_t* bp = btile + (size_t)(kk >> 5) * bstep;
    b0.q[0] = *(const uint4*)(bp);
    b0.q[1] = *(const uint4*)(bp + 8);
    b1.q[0] = *(const uint4*)(bp + 512);
    b1.q[1] = *(const uint4*)(bp + 520);
    acc0 = __builtin_amdgcn_wmma_f32_16x16x32_f16(false, a.v, false, b0.v,
                                                  (short)0, acc0, false, false);
    acc1 = __builtin_amdgcn_wmma_f32_16x16x32_f16(false, a.v, false, b1.v,
                                                  (short)0, acc1, false, false);
  }
  int c0 = nt2 * 32 + (lane & 15);
  int m0 = mt * 16 + (g << 3);
  float bv0 = bias ? bias[c0] : 0.0f;
  float bv1 = bias ? bias[c0 + 16] : 0.0f;
#pragma unroll
  for (int i = 0; i < 8; ++i) {
    C[(size_t)(m0 + i) * N + c0]      = acc0[i] + bv0;
    C[(size_t)(m0 + i) * N + c0 + 16] = acc1[i] + bv1;
  }
}

// ---------------------------------------------------------------------------
__global__ void k_zero(float* p, long n) {
  long i = (long)blockIdx.x * 256 + threadIdx.x;
  if (i < n) p[i] = 0.0f;
}

// f32 [K x N] row-major -> f16 fragment-swizzled B (rows >= K are zero-padded)
__global__ void k_swizB(const float* __restrict__ s, half_t* __restrict__ d,
                        int K, int N, int Kpad) {
  long i = (long)blockIdx.x * 256 + threadIdx.x;
  long total = (long)Kpad * N;
  if (i >= total) return;
  int e = (int)(i & 15);
  int lane = (int)((i >> 4) & 31);
  long tile = i >> 9;
  int ntiles = N >> 4;
  int nt = (int)(tile % ntiles);
  int kt = (int)(tile / ntiles);
  int krow = kt * 32 + (lane >> 4) * 8 + (e & 7) + (e >> 3) * 16;
  int col = nt * 16 + (lane & 15);
  d[i] = (krow < K) ? (half_t)s[(size_t)krow * N + col] : (half_t)0.0f;
}

// ---------------------------------------------------------------------------
// Edge embedding: bessel * poly cutoff, LN(8), GVP (8,1)->(32,1). Also edge counts.
// ---------------------------------------------------------------------------
__global__ void k_edge_embed(const float* __restrict__ pos, const int* __restrict__ src,
                             const int* __restrict__ dst,
                             const float* lng, const float* lnb, const float* wh,
                             const float* wsw, const float* wsb,
                             const float* wsvw, const float* wsvb, const float* wv,
                             float* __restrict__ hEs, float* __restrict__ hEv,
                             float* __restrict__ cnt, int E) {
  int e = blockIdx.x * blockDim.x + threadIdx.x;
  if (e >= E) return;
  int sj = src[e], di = dst[e];
  float vx = pos[sj * 3 + 0] - pos[di * 3 + 0];
  float vy = pos[sj * 3 + 1] - pos[di * 3 + 1];
  float vz = pos[sj * 3 + 2] - pos[di * 3 + 2];
  float len = sqrtf(vx * vx + vy * vy + vz * vz);
  float inv = 1.0f / len;
  float u = len * 0.1f;
  float u2 = u * u, u4 = u2 * u2, u5 = u4 * u, u6 = u5 * u, u7 = u6 * u;
  float env = (len < 10.0f) ? (1.0f - 21.0f * u5 + 35.0f * u6 - 15.0f * u7) : 0.0f;
  const float cb = 0.44721359549995794f;   // sqrt(2/10)
  float se[8];
  float mu = 0.0f;
#pragma unroll
  for (int k = 0; k < 8; ++k) {
    float arg = len * (float)(k + 1) * 0.31415926535897931f;  // n*pi/10
    se[k] = cb * __sinf(arg) * inv * env;
    mu += se[k];
  }
  mu *= 0.125f;
  float var = 0.0f;
#pragma unroll
  for (int k = 0; k < 8; ++k) { float d = se[k] - mu; var += d * d; }
  float rstd = rsqrtf(var * 0.125f + 1e-5f);
  float sn[8];
#pragma unroll
  for (int k = 0; k < 8; ++k) sn[k] = (se[k] - mu) * rstd * lng[k] + lnb[k];
  float w0 = wh[0];
  float hx = vx * inv * w0, hy = vy * inv * w0, hz = vz * inv * w0;
  float vn = sqrtf(fmaxf(hx * hx + hy * hy + hz * hz, WEPS));
  float gate = wsvb[0];
  for (int o = 0; o < 32; ++o) {
    float a = wsb[o] + vn * wsw[8 * 32 + o];
#pragma unroll
    for (int k = 0; k < 8; ++k) a += sn[k] * wsw[k * 32 + o];
    hEs[(size_t)e * 32 + o] = a;            // no relu (acts None)
    gate += a * wsvw[o];
  }
  float sg = sigm(gate) * wv[0];
  hEv[(size_t)e * 3 + 0] = hx * sg;
  hEv[(size_t)e * 3 + 1] = hy * sg;
  hEv[(size_t)e * 3 + 2] = hz * sg;
  atomicAdd(&cnt[di], 1.0f);
}

// ---------------------------------------------------------------------------
// Node embedding: hs = LN(emb[atom]) @ W + b ; hv = 0. One 128-thread block/node.
// ---------------------------------------------------------------------------
__global__ void k_node_embed(const int* __restrict__ atoms, const float* __restrict__ emb,
                             const float* lng, const float* lnb,
                             const float* __restrict__ W, const float* bias,
                             float* __restrict__ hs, float* __restrict__ hv, int N) {
  int n = blockIdx.x, t = threadIdx.x;
  __shared__ float red[128];
  __shared__ float sx[128];
  float x = emb[(size_t)atoms[n] * 128 + t];
  red[t] = x; __syncthreads();
  for (int s = 64; s > 0; s >>= 1) { if (t < s) red[t] += red[t + s]; __syncthreads(); }
  float mu = red[0] * (1.0f / 128.0f); __syncthreads();
  float d = x - mu;
  red[t] = d * d; __syncthreads();
  for (int s = 64; s > 0; s >>= 1) { if (t < s) red[t] += red[t + s]; __syncthreads(); }
  float rstd = rsqrtf(red[0] * (1.0f / 128.0f) + 1e-5f);
  sx[t] = d * rstd * lng[t] + lnb[t]; __syncthreads();
  float a = bias[t];
  for (int c = 0; c < 128; ++c) a += sx[c] * W[(size_t)c * 128 + t];
  hs[(size_t)n * 128 + t] = a;
  if (t < 48) hv[(size_t)n * 48 + t] = 0.0f;
}

// ---------------------------------------------------------------------------
// Message GVP0 prologue: gather + wh(33x33) + norms -> A0 rows (K=352) + vh0.
// ---------------------------------------------------------------------------
__global__ void k_msg_pre(const float* __restrict__ hs, const float* __restrict__ hv,
                          const float* __restrict__ hEs, const float* __restrict__ hEv,
                          const int* __restrict__ src, const int* __restrict__ dst,
                          const float* __restrict__ wh,
                          half_t* __restrict__ A0, float* __restrict__ vh_out,
                          int e0, int count) {
  __shared__ float mv[32][100];
  int t = threadIdx.x;
  int el = blockIdx.x * 32 + t;
  if (el >= count) return;
  int e = e0 + el;
  int sj = src[e], di = dst[e];
  half_t* arow = A0 + (size_t)el * 352;
  const float* s_src = hs + (size_t)sj * 128;
  const float* s_dst = hs + (size_t)di * 128;
  for (int c = 0; c < 128; ++c) arow[c]       = (half_t)s_src[c];
  for (int c = 0; c < 32;  ++c) arow[128 + c] = (half_t)hEs[(size_t)e * 32 + c];
  for (int c = 0; c < 128; ++c) arow[160 + c] = (half_t)s_dst[c];
  for (int c = 321; c < 352; ++c) arow[c] = (half_t)0.0f;
  const float* vs = hv + (size_t)sj * 48;
  const float* vd = hv + (size_t)di * 48;
  for (int i = 0; i < 48; ++i) mv[t][i] = vs[i];
  mv[t][48] = hEv[(size_t)e * 3 + 0];
  mv[t][49] = hEv[(size_t)e * 3 + 1];
  mv[t][50] = hEv[(size_t)e * 3 + 2];
  for (int i = 0; i < 48; ++i) mv[t][51 + i] = vd[i];
  float* vout = vh_out + (size_t)el * 99;
  for (int hh = 0; hh < 33; ++hh) {
    float ax = 0, ay = 0, az = 0;
    for (int vi = 0; vi < 33; ++vi) {
      float w = wh[vi * 33 + hh];
      ax += mv[t][vi * 3 + 0] * w;
      ay += mv[t][vi * 3 + 1] * w;
      az += mv[t][vi * 3 + 2] * w;
    }
    vout[hh * 3 + 0] = ax; vout[hh * 3 + 1] = ay; vout[hh * 3 + 2] = az;
    arow[288 + hh] = (half_t)sqrtf(fmaxf(ax * ax + ay * ay + az * az, WEPS));
  }
}

// ---------------------------------------------------------------------------
// Message GVP0 epilogue + GVP1 prologue (h_in = 33).
// ---------------------------------------------------------------------------
__global__ void k_msg_mid0(const float* __restrict__ S, const float* __restrict__ vh_in,
                           const float* __restrict__ wv, const float* __restrict__ wsvw,
                           const float* wsvb, const float* __restrict__ wh_next,
                           half_t* __restrict__ A_out, float* __restrict__ vh_out, int count) {
  __shared__ float sv[32][100];
  __shared__ float ss[32][128];
  int t = threadIdx.x;
  int el = blockIdx.x * 32 + t;
  if (el >= count) return;
  for (int i = 0; i < 99;  ++i) sv[t][i] = vh_in[(size_t)el * 99 + i];
  for (int c = 0; c < 128; ++c) ss[t][c] = S[(size_t)el * 128 + c];
  float vo[48];
  for (int o = 0; o < 16; ++o) {
    float gt = wsvb[o];
    for (int c = 0; c < 128; ++c) gt += ss[t][c] * wsvw[c * 16 + o];
    float sg = sigm(gt);
#pragma unroll
    for (int c3 = 0; c3 < 3; ++c3) {
      float a = 0;
      for (int h = 0; h < 33; ++h) a += sv[t][h * 3 + c3] * wv[h * 16 + o];
      vo[o * 3 + c3] = a * sg;
    }
  }
  half_t* arow = A_out + (size_t)el * 160;
  for (int c = 0; c < 128; ++c) arow[c] = (half_t)fmaxf(ss[t][c], 0.0f);   // relu
  float* vout = vh_out + (size_t)el * 48;
  for (int hh = 0; hh < 16; ++hh) {
    float ax = 0, ay = 0, az = 0;
#pragma unroll
    for (int o = 0; o < 16; ++o) {
      float w = wh_next[o * 16 + hh];
      ax += vo[o * 3 + 0] * w; ay += vo[o * 3 + 1] * w; az += vo[o * 3 + 2] * w;
    }
    vout[hh * 3 + 0] = ax; vout[hh * 3 + 1] = ay; vout[hh * 3 + 2] = az;
    arow[128 + hh] = (half_t)sqrtf(fmaxf(ax * ax + ay * ay + az * az, WEPS));
  }
  for (int c = 144; c < 160; ++c) arow[c] = (half_t)0.0f;
}

// ---------------------------------------------------------------------------
// Message GVP1 epilogue + GVP2 prologue (h_in = 16, register-resident).
// ---------------------------------------------------------------------------
__global__ void k_msg_mid1(const float* __restrict__ S, const float* __restrict__ vh_in,
                           const float* __restrict__ wv, const float* __restrict__ wsvw,
                           const float* wsvb, const float* __restrict__ wh_next,
                           half_t* __restrict__ A_out, float* __restrict__ vh_out, int count) {
  int el = blockIdx.x * blockDim.x + threadIdx.x;
  if (el >= count) return;
  float vh[48];
  const float* vp = vh_in + (size_t)el * 48;
#pragma unroll
  for (int i = 0; i < 48; ++i) vh[i] = vp[i];
  const float* s = S + (size_t)el * 128;
  float vo[48];
  for (int o = 0; o < 16; ++o) {
    float gt = wsvb[o];
    for (int c = 0; c < 128; ++c) gt += s[c] * wsvw[c * 16 + o];
    float sg = sigm(gt);
#pragma unroll
    for (int c3 = 0; c3 < 3; ++c3) {
      float a = 0;
#pragma unroll
      for (int h = 0; h < 16; ++h) a += vh[h * 3 + c3] * wv[h * 16 + o];
      vo[o * 3 + c3] = a * sg;
    }
  }
  half_t* arow = A_out + (size_t)el * 160;
  for (int c = 0; c < 128; ++c) arow[c] = (half_t)fmaxf(s[c], 0.0f);
  float* vout = vh_out + (size_t)el * 48;
  for (int hh = 0; hh < 16; ++hh) {
    float ax = 0, ay = 0, az = 0;
#pragma unroll
    for (int o = 0; o < 16; ++o) {
      float w = wh_next[o * 16 + hh];
      ax += vo[o * 3 + 0] * w; ay += vo[o * 3 + 1] * w; az += vo[o * 3 + 2] * w;
    }
    vout[hh * 3 + 0] = ax; vout[hh * 3 + 1] = ay; vout[hh * 3 + 2] = az;
    arow[128 + hh] = (half_t)sqrtf(fmaxf(ax * ax + ay * ay + az * az, WEPS));
  }
  for (int c = 144; c < 160; ++c) arow[c] = (half_t)0.0f;
}

// ---------------------------------------------------------------------------
// Message GVP2 epilogue + scatter (mean-aggregation numerator via atomics).
// ---------------------------------------------------------------------------
__global__ void k_msg_post(const float* __restrict__ S, const float* __restrict__ vh_in,
                           const float* __restrict__ wv, const float* __restrict__ wsvw,
                           const float* wsvb, const int* __restrict__ dst,
                           float* __restrict__ ds, float* __restrict__ dv,
                           int e0, int count) {
  int el = blockIdx.x * blockDim.x + threadIdx.x;
  if (el >= count) return;
  int di = dst[e0 + el];
  float vh[48];
  const float* vp = vh_in + (size_t)el * 48;
#pragma unroll
  for (int i = 0; i < 48; ++i) vh[i] = vp[i];
  const float* s = S + (size_t)el * 128;
  for (int o = 0; o < 16; ++o) {
    float gt = wsvb[o];
    for (int c = 0; c < 128; ++c) gt += s[c] * wsvw[c * 16 + o];
    float sg = sigm(gt);
#pragma unroll
    for (int c3 = 0; c3 < 3; ++c3) {
      float a = 0;
#pragma unroll
      for (int h = 0; h < 16; ++h) a += vh[h * 3 + c3] * wv[h * 16 + o];
      atomicAdd(&dv[(size_t)di * 48 + o * 3 + c3], a * sg);
    }
  }
  for (int c = 0; c < 128; ++c) atomicAdd(&ds[(size_t)di * 128 + c], s[c]); // no relu
}

// ---------------------------------------------------------------------------
// Residual + LayerNorm (scalar) + vector RMS-norm. One warp (32 lanes) per node.
// cnt == nullptr -> divisor 1.
// ---------------------------------------------------------------------------
__global__ void k_update(float* __restrict__ hs, float* __restrict__ hv,
                         const float* __restrict__ ds, const float* __restrict__ dv,
                         const float* cnt, const float* lg, const float* lb, int N) {
  int gid = blockIdx.x * blockDim.x + threadIdx.x;
  int lane = gid & 31;
  int n = gid >> 5;
  if (n >= N) return;
  float ic = cnt ? 1.0f / fmaxf(cnt[n], 1.0f) : 1.0f;
  float x[4];
  float s = 0.0f;
#pragma unroll
  for (int j = 0; j < 4; ++j) {
    int c = lane * 4 + j;
    x[j] = hs[(size_t)n * 128 + c] + ds[(size_t)n * 128 + c] * ic;
    s += x[j];
  }
  for (int off = 16; off; off >>= 1) s += __shfl_xor(s, off, 32);
  float mu = s * (1.0f / 128.0f);
  float vv = 0.0f;
#pragma unroll
  for (int j = 0; j < 4; ++j) { float d = x[j] - mu; vv += d * d; }
  for (int off = 16; off; off >>= 1) vv += __shfl_xor(vv, off, 32);
  float rstd = rsqrtf(vv * (1.0f / 128.0f) + 1e-5f);
#pragma unroll
  for (int j = 0; j < 4; ++j) {
    int c = lane * 4 + j;
    hs[(size_t)n * 128 + c] = (x[j] - mu) * rstd * lg[c] + lb[c];
  }
  float vx = 0, vy = 0, vz = 0, n2 = 0;
  if (lane < 16) {
    size_t b = (size_t)n * 48 + lane * 3;
    vx = hv[b + 0] + dv[b + 0] * ic;
    vy = hv[b + 1] + dv[b + 1] * ic;
    vz = hv[b + 2] + dv[b + 2] * ic;
    n2 = fmaxf(vx * vx + vy * vy + vz * vz, WEPS);
  }
  float tsum = n2;
  for (int off = 16; off; off >>= 1) tsum += __shfl_xor(tsum, off, 32);
  float rr = rsqrtf(tsum * (1.0f / 16.0f));
  if (lane < 16) {
    size_t b = (size_t)n * 48 + lane * 3;
    hv[b + 0] = vx * rr; hv[b + 1] = vy * rr; hv[b + 2] = vz * rr;
  }
}

// ---------------------------------------------------------------------------
// Feed-forward GVP0 prologue (h=32).
// ---------------------------------------------------------------------------
__global__ void k_ff_pre(const float* __restrict__ hs, const float* __restrict__ hv,
                         const float* __restrict__ wh,
                         half_t* __restrict__ A, float* __restrict__ vh_out, int N) {
  int n = blockIdx.x * blockDim.x + threadIdx.x;
  if (n >= N) return;
  float v[48];
#pragma unroll
  for (int i = 0; i < 48; ++i) v[i] = hv[(size_t)n * 48 + i];
  half_t* arow = A + (size_t)n * 160;
  const float* s = hs + (size_t)n * 128;
  for (int c = 0; c < 128; ++c) arow[c] = (half_t)s[c];
  for (int hh = 0; hh < 32; ++hh) {
    float ax = 0, ay = 0, az = 0;
#pragma unroll
    for (int vi = 0; vi < 16; ++vi) {
      float w = wh[vi * 32 + hh];
      ax += v[vi * 3 + 0] * w; ay += v[vi * 3 + 1] * w; az += v[vi * 3 + 2] * w;
    }
    float* vo = vh_out + (size_t)n * 96 + hh * 3;
    vo[0] = ax; vo[1] = ay; vo[2] = az;
    arow[128 + hh] = (half_t)sqrtf(fmaxf(ax * ax + ay * ay + az * az, WEPS));
  }
}

// ---------------------------------------------------------------------------
// FF GVP0 epilogue + GVP1 prologue (h=32 both sides, LDS staged).
// ---------------------------------------------------------------------------
__global__ void k_ff_mid(const float* __restrict__ F0, const float* __restrict__ vh_in,
                         const float* __restrict__ wv, const float* __restrict__ wsvw,
                         const float* wsvb, const float* __restrict__ wh_next,
                         half_t* __restrict__ A, float* __restrict__ vh_out, int N) {
  __shared__ float sv[32][96];
  __shared__ float so[32][96];
  int t = threadIdx.x;
  int n = blockIdx.x * 32 + t;
  if (n >= N) return;
  for (int i = 0; i < 96; ++i) sv[t][i] = vh_in[(size_t)n * 96 + i];
  const float* f0 = F0 + (size_t)n * 512;
  for (int o = 0; o < 32; ++o) {
    float gt = wsvb[o];
    for (int c = 0; c < 512; ++c) gt += f0[c] * wsvw[c * 32 + o];
    float sg = sigm(gt);
#pragma unroll
    for (int c3 = 0; c3 < 3; ++c3) {
      float a = 0;
      for (int h = 0; h < 32; ++h) a += sv[t][h * 3 + c3] * wv[h * 32 + o];
      so[t][o * 3 + c3] = a * sg;
    }
  }
  half_t* arow = A + (size_t)n * 544;
  for (int c = 0; c < 512; ++c) arow[c] = (half_t)fmaxf(f0[c], 0.0f);   // relu
  for (int hh = 0; hh < 32; ++hh) {
    float ax = 0, ay = 0, az = 0;
    for (int o = 0; o < 32; ++o) {
      float w = wh_next[o * 32 + hh];
      ax += so[t][o * 3 + 0] * w; ay += so[t][o * 3 + 1] * w; az += so[t][o * 3 + 2] * w;
    }
    float* vo = vh_out + (size_t)n * 96 + hh * 3;
    vo[0] = ax; vo[1] = ay; vo[2] = az;
    arow[512 + hh] = (half_t)sqrtf(fmaxf(ax * ax + ay * ay + az * az, WEPS));
  }
}

// ---------------------------------------------------------------------------
// FF GVP1 epilogue: gated vector output (no relu on scalars).
// ---------------------------------------------------------------------------
__global__ void k_ff_post(const float* __restrict__ F1, const float* __restrict__ vh_in,
                          const float* __restrict__ wv, const float* __restrict__ wsvw,
                          const float* wsvb, float* __restrict__ dvf, int N) {
  __shared__ float sv[32][96];
  int t = threadIdx.x;
  int n = blockIdx.x * 32 + t;
  if (n >= N) return;
  for (int i = 0; i < 96; ++i) sv[t][i] = vh_in[(size_t)n * 96 + i];
  const float* f1 = F1 + (size_t)n * 128;
  for (int o = 0; o < 16; ++o) {
    float gt = wsvb[o];
    for (int c = 0; c < 128; ++c) gt += f1[c] * wsvw[c * 16 + o];
    float sg = sigm(gt);
#pragma unroll
    for (int c3 = 0; c3 < 3; ++c3) {
      float a = 0;
      for (int h = 0; h < 32; ++h) a += sv[t][h * 3 + c3] * wv[h * 16 + o];
      dvf[(size_t)n * 48 + o * 3 + c3] = a * sg;
    }
  }
}

// ---------------------------------------------------------------------------
__global__ void k_pool(const float* __restrict__ hs, const int* __restrict__ batch,
                       float* __restrict__ pooled, int N) {
  long i = (long)blockIdx.x * 256 + threadIdx.x;
  if (i >= (long)N * 128) return;
  int n = (int)(i >> 7);
  int c = (int)(i & 127);
  atomicAdd(&pooled[(size_t)batch[n] * 128 + c], hs[i]);
}

__global__ void k_pred(const float* __restrict__ pooled, const float* __restrict__ w1,
                       const float* b1, const float* __restrict__ w2, const float* b2,
                       float* __restrict__ out) {
  __shared__ float h[128];
  int g = blockIdx.x, t = threadIdx.x;
  const float* p = pooled + (size_t)g * 128;
  float a = b1[t];
  for (int c = 0; c < 128; ++c) a += p[c] * w1[(size_t)c * 128 + t];
  h[t] = fmaxf(a, 0.0f);
  __syncthreads();
  if (t == 0) {
    float o = b2[0];
    for (int c = 0; c < 128; ++c) o += h[c] * w2[c];
    out[g] = o;
  }
}

// ---------------------------------------------------------------------------
extern "C" void kernel_launch(void* const* d_in, const int* in_sizes, int n_in,
                              void* d_out, int out_size, void* d_ws, size_t ws_size,
                              hipStream_t stream) {
  (void)n_in; (void)ws_size;
  // pytree (sorted-key) flatten order of setup_inputs():
  // 0 atoms, 1 batch_ids, 2 edge_index, 3..189 params leaves, 190 pos
  const int* atoms = (const int*)d_in[0];
  const int* batch = (const int*)d_in[1];
  const int* eidx  = (const int*)d_in[2];
  const float* pos = (const float*)d_in[190];
  auto F = [&](int i) -> const float* { return (const float*)d_in[i]; };

  int N = in_sizes[0];
  int E = in_sizes[2] / 2;
  int G = out_size;
  const int* srcI = eidx;
  const int* dstI = eidx + E;
  int Np = (N + 15) & ~15;

  // ---- workspace carve ----
  char* wp = (char*)d_ws;
  auto alloc = [&](size_t bytes) -> void* {
    void* p = (void*)wp;
    wp += (bytes + 255) & ~(size_t)255;
    return p;
  };
  float* hs   = (float*)alloc((size_t)N * 128 * 4);
  float* hv   = (float*)alloc((size_t)N * 48 * 4);
  float* dsb  = (float*)alloc((size_t)N * 128 * 4);
  float* dvb  = (float*)alloc((size_t)N * 48 * 4);
  float* cnt  = (float*)alloc((size_t)N * 4);
  float* hEs  = (float*)alloc((size_t)E * 32 * 4);
  float* hEv  = (float*)alloc((size_t)E * 3 * 4);
  float* pooled = (float*)alloc((size_t)G * 128 * 4);
  half_t *Wm0[5], *Wm1[5], *Wm2[5], *Wf0[5], *Wf1[5];
  for (int l = 0; l < 5; ++l) {
    Wm0[l] = (half_t*)alloc((size_t)352 * 128 * 2);
    Wm1[l] = (half_t*)alloc((size_t)160 * 128 * 2);
    Wm2[l] = (half_t*)alloc((size_t)160 * 128 * 2);
    Wf0[l] = (half_t*)alloc((size_t)160 * 512 * 2);
    Wf1[l] = (half_t*)alloc((size_t)544 * 128 * 2);
  }
  const int EC = 20000;  // edge chunk (multiple of 16 and 32)
  half_t* A0 = (half_t*)alloc((size_t)EC * 352 * 2);
  half_t* A1 = (half_t*)alloc((size_t)EC * 160 * 2);   // reused for A2
  float* Sa  = (float*)alloc((size_t)EC * 128 * 4);
  float* Sb  = (float*)alloc((size_t)EC * 128 * 4);
  float* vh0 = (float*)alloc((size_t)EC * 99 * 4);
  float* vhA = (float*)alloc((size_t)EC * 48 * 4);
  float* vhB = (float*)alloc((size_t)EC * 48 * 4);
  half_t* Af0 = (half_t*)alloc((size_t)Np * 160 * 2);
  float* F0b  = (float*)alloc((size_t)Np * 512 * 4);
  half_t* Af1 = (half_t*)alloc((size_t)Np * 544 * 2);
  float* F1b  = (float*)alloc((size_t)Np * 128 * 4);
  float* vfA  = (float*)alloc((size_t)N * 96 * 4);
  float* vfB  = (float*)alloc((size_t)N * 96 * 4);
  float* dvf  = (float*)alloc((size_t)N * 48 * 4);

  auto conv = [&](const float* s, half_t* d, int K, int Nn, int Kpad) {
    long tot = (long)Kpad * Nn;
    k_swizB<<<(unsigned)((tot + 255) / 256), 256, 0, stream>>>(s, d, K, Nn, Kpad);
  };
  auto gemm = [&](const half_t* A, const half_t* B, const float* bias, float* C,
                  int M, int Nn, int K) {
    dim3 grid(Nn / 32, (M / 16 + 7) / 8);
    k_gemm<<<grid, 256, 0, stream>>>(A, B, bias, C, M, Nn, K);
  };

  // ---- weight conversion to f16 fragment-swizzled B (K padded to mult-of-32) ----
  for (int l = 0; l < 5; ++l) {
    int LB = 16 + 34 * l;
    conv(F(LB + 18), Wm0[l], 321, 128, 352);   // msg0.ws.w
    conv(F(LB + 24), Wm1[l], 144, 128, 160);   // msg1.ws.w
    conv(F(LB + 30), Wm2[l], 144, 128, 160);   // msg2.ws.w
    conv(F(LB + 2),  Wf0[l], 160, 512, 160);   // ff0.ws.w
    conv(F(LB + 8),  Wf1[l], 544, 128, 544);   // ff1.ws.w
  }

  // ---- embeddings ----
  k_zero<<<(N + 255) / 256, 256, 0, stream>>>(cnt, (long)N);
  k_edge_embed<<<(E + 255) / 256, 256, 0, stream>>>(
      pos, srcI, dstI, F(10), F(9), F(3), F(5), F(4), F(7), F(6), F(8),
      hEs, hEv, cnt, E);
  k_node_embed<<<N, 128, 0, stream>>>(atoms, F(15), F(12), F(11), F(14), F(13), hs, hv, N);

  // ---- 5 GVPConv layers ----
  for (int l = 0; l < 5; ++l) {
    int LB = 16 + 34 * l;
    k_zero<<<(unsigned)(((long)N * 128 + 255) / 256), 256, 0, stream>>>(dsb, (long)N * 128);
    k_zero<<<(unsigned)(((long)N * 48 + 255) / 256), 256, 0, stream>>>(dvb, (long)N * 48);
    for (int e0 = 0; e0 < E; e0 += EC) {
      int cE = (E - e0 < EC) ? (E - e0) : EC;
      int Mp = (cE + 15) & ~15;
      k_msg_pre<<<(cE + 31) / 32, 32, 0, stream>>>(hs, hv, hEs, hEv, srcI, dstI,
                                                   F(LB + 16), A0, vh0, e0, cE);
      gemm(A0, Wm0[l], F(LB + 17), Sa, Mp, 128, 352);
      k_msg_mid0<<<(cE + 31) / 32, 32, 0, stream>>>(Sa, vh0, F(LB + 21), F(LB + 20),
                                                    F(LB + 19), F(LB + 22), A1, vhA, cE);
      gemm(A1, Wm1[l], F(LB + 23), Sb, Mp, 128, 160);
      k_msg_mid1<<<(cE + 127) / 128, 128, 0, stream>>>(Sb, vhA, F(LB + 27), F(LB + 26),
                                                       F(LB + 25), F(LB + 28), A1, vhB, cE);
      gemm(A1, Wm2[l], F(LB + 29), Sa, Mp, 128, 160);
      k_msg_post<<<(cE + 127) / 128, 128, 0, stream>>>(Sa, vhB, F(LB + 33), F(LB + 32),
                                                       F(LB + 31), dstI, dsb, dvb, e0, cE);
    }
    k_update<<<(unsigned)(((long)N * 32 + 255) / 256), 256, 0, stream>>>(
        hs, hv, dsb, dvb, cnt, F(LB + 13), F(LB + 12), N);
    // feed-forward
    k_ff_pre<<<(N + 127) / 128, 128, 0, stream>>>(hs, hv, F(LB + 0), Af0, vfA, N);
    gemm(Af0, Wf0[l], F(LB + 1), F0b, Np, 512, 160);
    k_ff_mid<<<(N + 31) / 32, 32, 0, stream>>>(F0b, vfA, F(LB + 5), F(LB + 4),
                                               F(LB + 3), F(LB + 6), Af1, vfB, N);
    gemm(Af1, Wf1[l], F(LB + 7), F1b, Np, 128, 544);
    k_ff_post<<<(N + 31) / 32, 32, 0, stream>>>(F1b, vfB, F(LB + 11), F(LB + 10),
                                                F(LB + 9), dvf, N);
    k_update<<<(unsigned)(((long)N * 32 + 255) / 256), 256, 0, stream>>>(
        hs, hv, F1b, dvf, (const float*)nullptr, F(LB + 15), F(LB + 14), N);
  }

  // ---- pool + prediction MLP ----
  k_zero<<<((G * 128) + 255) / 256, 256, 0, stream>>>(pooled, (long)G * 128);
  k_pool<<<(unsigned)(((long)N * 128 + 255) / 256), 256, 0, stream>>>(hs, batch, pooled, N);
  k_pred<<<G, 128, 0, stream>>>(pooled, F(187), F(186), F(189), F(188), (float*)d_out);
}
